// MultiHeadsLinearAttention_41815801594399
// MI455X (gfx1250) — matrix operand order repaired and checked by reading, not compiled, not run on hardware
//
#include <hip/hip_runtime.h>

#define DEVI __device__ __forceinline__

typedef __attribute__((ext_vector_type(16))) __bf16 v16bf;
typedef __attribute__((ext_vector_type(8)))  float  v8f;

constexpr int NB   = 8;     // batch
constexpr int CC   = 512;   // channels
constexpr int NN   = 4096;  // pixels (64x64)
constexpr int NH   = 8;     // heads
constexpr int CHD  = 64;    // channels per head
constexpr float SQRT_C = 22.627416997969522f; // sqrt(512)

// ---------------- WMMA fragment helpers (CDNA5 wave32 layouts) ----------------

union FragU { v16bf v; uint4 q[2]; };

// A matrix 16x32 bf16, memory row-major [M][K] with row stride ld.
DEVI v16bf loadA(const __bf16* base, int ld, int row, int kb, int lane) {
  const __bf16* p = base + (size_t)(row + (lane & 15)) * ld + kb + ((lane >> 4) << 3);
  FragU u;
  u.q[0] = *(const uint4*)(p);
  u.q[1] = *(const uint4*)(p + 16);
  return u.v;
}

// B matrix 32x16 bf16, memory laid out [N-column][K] contiguous in K, stride ld.
DEVI v16bf loadB(const __bf16* base, int ld, int col, int kb, int lane) {
  const __bf16* p = base + (size_t)(col + (lane & 15)) * ld + kb + ((lane >> 4) << 3);
  FragU u;
  u.q[0] = *(const uint4*)(p);
  u.q[1] = *(const uint4*)(p + 16);
  return u.v;
}

DEVI v8f wmma_bf16(v16bf a, v16bf b, v8f c) {
  return __builtin_amdgcn_wmma_f32_16x16x32_bf16(false, a, false, b, (short)0, c, false, false);
}

// D 16x16 f32 -> bf16 store: lane holds column col+(lane&15), rows r0..r0+7
DEVI void storeD_bf16(__bf16* base, int ld, int row, int col, v8f acc, int lane) {
  int r0 = row + ((lane >> 4) << 3);
  int c  = col + (lane & 15);
  #pragma unroll
  for (int r = 0; r < 8; ++r)
    base[(size_t)(r0 + r) * ld + c] = (__bf16)acc[r];
}

DEVI float wave_sum(float v) {
  #pragma unroll
  for (int m = 16; m; m >>= 1) v += __shfl_xor(v, m, 32);
  return v;
}
DEVI float wave_max(float v) {
  #pragma unroll
  for (int m = 16; m; m >>= 1) v = fmaxf(v, __shfl_xor(v, m, 32));
  return v;
}

// Software-pipelined 16xK * Kx64 tile GEMM: double-buffered fragments so the
// next k-step's global_load_b128s are in flight while the current WMMAs run.
template<int KTOT>
DEVI void gemm16x64(const __bf16* __restrict__ Abase, int lda, int row,
                    const __bf16* __restrict__ Bbase, int ldb, int col,
                    int lane, v8f acc[4]) {
  v16bf a  = loadA(Abase, lda, row, 0, lane);
  v16bf b0 = loadB(Bbase, ldb, col +  0, 0, lane);
  v16bf b1 = loadB(Bbase, ldb, col + 16, 0, lane);
  v16bf b2 = loadB(Bbase, ldb, col + 32, 0, lane);
  v16bf b3 = loadB(Bbase, ldb, col + 48, 0, lane);
  for (int kb = 32; kb < KTOT; kb += 32) {
    v16bf an  = loadA(Abase, lda, row, kb, lane);
    v16bf bn0 = loadB(Bbase, ldb, col +  0, kb, lane);
    v16bf bn1 = loadB(Bbase, ldb, col + 16, kb, lane);
    v16bf bn2 = loadB(Bbase, ldb, col + 32, kb, lane);
    v16bf bn3 = loadB(Bbase, ldb, col + 48, kb, lane);
    acc[0] = wmma_bf16(a, b0, acc[0]);
    acc[1] = wmma_bf16(a, b1, acc[1]);
    acc[2] = wmma_bf16(a, b2, acc[2]);
    acc[3] = wmma_bf16(a, b3, acc[3]);
    a = an; b0 = bn0; b1 = bn1; b2 = bn2; b3 = bn3;
  }
  acc[0] = wmma_bf16(a, b0, acc[0]);
  acc[1] = wmma_bf16(a, b1, acc[1]);
  acc[2] = wmma_bf16(a, b2, acc[2]);
  acc[3] = wmma_bf16(a, b3, acc[3]);
}

// ---------------- Kernel 0: weights fp32 -> bf16 ----------------
__global__ void k_cvt_w(const float* __restrict__ wq, const float* __restrict__ wf,
                        __bf16* __restrict__ dq, __bf16* __restrict__ df) {
  int i = blockIdx.x * blockDim.x + threadIdx.x;
  if (i < 3 * CC * CC) dq[i] = (__bf16)wq[i];
  if (i < CC * CC)     df[i] = (__bf16)wf[i];
}

// ---------------- Kernel 1: RMSNorm over C + transpose -> xn[b][n][c] bf16 ----------------
__global__ void k_rmsnorm_t(const float* __restrict__ x, const float* __restrict__ g,
                            __bf16* __restrict__ xn) {
  int wave = (blockIdx.x * blockDim.x + threadIdx.x) >> 5;   // one wave per pixel
  int lane = threadIdx.x & 31;
  int b = wave >> 12, n = wave & (NN - 1);
  const float* xp = x + (size_t)b * CC * NN + n;
  int c0 = lane * 16;
  float vals[16]; float ss = 0.f;
  #pragma unroll
  for (int j = 0; j < 16; ++j) {
    float v = xp[(size_t)(c0 + j) * NN];
    vals[j] = v; ss += v * v;
  }
  ss = wave_sum(ss);
  float sc = SQRT_C / fmaxf(sqrtf(ss), 1e-12f);
  union { uint4 q[2]; __bf16 e[16]; } o;
  #pragma unroll
  for (int j = 0; j < 16; ++j) o.e[j] = (__bf16)(vals[j] * sc * g[c0 + j]);
  uint4* dst = (uint4*)(xn + ((size_t)b * NN + n) * CC + c0);
  dst[0] = o.q[0]; dst[1] = o.q[1];
}

// ---------------- Kernel 2a: q GEMM  D[n][o] = xn[n][c] * Wq[o][c],  o in [0,512) ----------------
__global__ void k_gemm_q(const __bf16* __restrict__ xn, const __bf16* __restrict__ wq,
                         __bf16* __restrict__ qraw) {
  int wave = (blockIdx.x * blockDim.x + threadIdx.x) >> 5;
  int lane = threadIdx.x & 31;
  int b = wave >> 11;                 // 2048 waves/batch
  int rest = wave & 2047;
  int n_tile = (rest >> 3) << 4;      // 256 row tiles
  int o_base = (rest & 7) << 6;       // 8 groups of 64 cols
  const __bf16* A = xn + (size_t)b * NN * CC;
  v8f acc[4] = {{}, {}, {}, {}};
  gemm16x64<CC>(A, CC, n_tile, wq, CC, o_base, lane, acc);
  __bf16* D = qraw + (size_t)b * NN * CC;
  storeD_bf16(D, CC, n_tile, o_base +  0, acc[0], lane);
  storeD_bf16(D, CC, n_tile, o_base + 16, acc[1], lane);
  storeD_bf16(D, CC, n_tile, o_base + 32, acc[2], lane);
  storeD_bf16(D, CC, n_tile, o_base + 48, acc[3], lane);
}

// ---------------- Kernel 2b: k/v GEMM  D[o2][n] = W[512+o2][c] * xn[n][c] ----------------
__global__ void k_gemm_kv(const __bf16* __restrict__ xn, const __bf16* __restrict__ wq,
                          __bf16* __restrict__ kv) {
  int wave = (blockIdx.x * blockDim.x + threadIdx.x) >> 5;
  int lane = threadIdx.x & 31;
  int b = wave >> 12;                 // 4096 waves/batch
  int rest = wave & 4095;
  int o_tile = (rest >> 6) << 4;      // 64 row tiles (1024 k/v rows)
  int n_base = (rest & 63) << 6;      // 64 groups of 64 cols
  const __bf16* A  = wq + (size_t)CC * CC;       // rows 512..1535 of qkv_w
  const __bf16* Bm = xn + (size_t)b * NN * CC;
  v8f acc[4] = {{}, {}, {}, {}};
  gemm16x64<CC>(A, CC, o_tile, Bm, CC, n_base, lane, acc);
  __bf16* D = kv + (size_t)b * 1024 * NN;
  storeD_bf16(D, NN, o_tile, n_base +  0, acc[0], lane);
  storeD_bf16(D, NN, o_tile, n_base + 16, acc[1], lane);
  storeD_bf16(D, NN, o_tile, n_base + 32, acc[2], lane);
  storeD_bf16(D, NN, o_tile, n_base + 48, acc[3], lane);
}

// ---------------- Kernel 3: q softmax over channel (64) * scale, in place ----------------
__global__ void k_softmax_q(__bf16* __restrict__ qraw) {
  int wave = (blockIdx.x * blockDim.x + threadIdx.x) >> 5;  // one wave per (b,n,h)
  int lane = threadIdx.x & 31;
  int h = wave & 7;
  int n = (wave >> 3) & (NN - 1);
  int b = wave >> 15;
  __bf16* p = qraw + ((size_t)b * NN + n) * CC + h * CHD + lane * 2;
  float e0 = (float)p[0], e1 = (float)p[1];
  float m = wave_max(fmaxf(e0, e1));
  e0 = __expf(e0 - m); e1 = __expf(e1 - m);
  float s = wave_sum(e0 + e1);
  float sc = 0.125f / s;                       // * CH^-0.5
  p[0] = (__bf16)(e0 * sc);
  p[1] = (__bf16)(e1 * sc);
}

// ---------------- Kernel 4: k softmax over n (4096), in place; one block per (b,o) ----------------
__global__ void k_softmax_k(__bf16* __restrict__ kv) {
  __shared__ float red[256];
  int row = blockIdx.x;              // b*512 + o, o < 512 (k rows)
  int b = row >> 9, o = row & 511;
  __bf16* p = kv + ((size_t)b * 1024 + o) * NN;
  int t = threadIdx.x;
  float vals[16];
  float m = -3.4e38f;
  #pragma unroll
  for (int j = 0; j < 16; ++j) { float v = (float)p[t + j * 256]; vals[j] = v; m = fmaxf(m, v); }
  red[t] = m; __syncthreads();
  for (int s = 128; s; s >>= 1) { if (t < s) red[t] = fmaxf(red[t], red[t + s]); __syncthreads(); }
  m = red[0]; __syncthreads();
  float sum = 0.f;
  #pragma unroll
  for (int j = 0; j < 16; ++j) { vals[j] = __expf(vals[j] - m); sum += vals[j]; }
  red[t] = sum; __syncthreads();
  for (int s = 128; s; s >>= 1) { if (t < s) red[t] += red[t + s]; __syncthreads(); }
  float inv = 1.0f / red[0];
  #pragma unroll
  for (int j = 0; j < 16; ++j) p[t + j * 256] = (__bf16)(vals[j] * inv);
}

// ---------------- Kernel 5: attn[c][d] = sum_n k[c][n] v[d][n]; store transposed [d][c] ----------------
__global__ void k_attn(const __bf16* __restrict__ kv, __bf16* __restrict__ attn_t) {
  int bh = blockIdx.x;               // 64 blocks, 16 waves each
  int b = bh >> 3, h = bh & 7;
  int w = threadIdx.x >> 5, lane = threadIdx.x & 31;
  int c_tile = (w >> 2) << 4, d_tile = (w & 3) << 4;
  const __bf16* K = kv + ((size_t)b * 1024 + h * CHD) * NN;
  const __bf16* V = kv + ((size_t)b * 1024 + 512 + h * CHD) * NN;
  v8f acc = {};
  // double-buffered single-tile K-loop
  v16bf a  = loadA(K, NN, c_tile, 0, lane);
  v16bf bf = loadB(V, NN, d_tile, 0, lane);
  for (int kb = 32; kb < NN; kb += 32) {
    v16bf an  = loadA(K, NN, c_tile, kb, lane);
    v16bf bfn = loadB(V, NN, d_tile, kb, lane);
    acc = wmma_bf16(a, bf, acc);
    a = an; bf = bfn;
  }
  acc = wmma_bf16(a, bf, acc);
  __bf16* T = attn_t + (size_t)bh * CHD * CHD;  // [d][c]
  int d  = d_tile + (lane & 15);
  int c0 = c_tile + ((lane >> 4) << 3);
  union { uint4 q; __bf16 e[8]; } u;
  #pragma unroll
  for (int r = 0; r < 8; ++r) u.e[r] = (__bf16)acc[r];
  *(uint4*)(T + (size_t)d * CHD + c0) = u.q;
}

// ---------------- Kernel 6: out2[n][h*64+d] = sum_c q[n][c] attn[c][d] ----------------
__global__ void k_out(const __bf16* __restrict__ qs, const __bf16* __restrict__ attn_t,
                      __bf16* __restrict__ out2) {
  int wave = (blockIdx.x * blockDim.x + threadIdx.x) >> 5;
  int lane = threadIdx.x & 31;
  int n_tile = (wave & 255) << 4;
  int h = (wave >> 8) & 7;
  int b = wave >> 11;
  const __bf16* A = qs + (size_t)b * NN * CC + h * CHD;        // ld=CC, K = 0..63
  const __bf16* T = attn_t + (size_t)(b * 8 + h) * CHD * CHD;  // [d][c], ld=CHD
  // K=64 -> 2 steps; hoist ALL loads, then 8 back-to-back WMMAs
  v16bf a0 = loadA(A, CC, n_tile, 0, lane);
  v16bf a1 = loadA(A, CC, n_tile, 32, lane);
  v16bf c00 = loadB(T, CHD,  0, 0, lane), c01 = loadB(T, CHD,  0, 32, lane);
  v16bf c10 = loadB(T, CHD, 16, 0, lane), c11 = loadB(T, CHD, 16, 32, lane);
  v16bf c20 = loadB(T, CHD, 32, 0, lane), c21 = loadB(T, CHD, 32, 32, lane);
  v16bf c30 = loadB(T, CHD, 48, 0, lane), c31 = loadB(T, CHD, 48, 32, lane);
  v8f r0 = {}, r1 = {}, r2 = {}, r3 = {};
  r0 = wmma_bf16(a0, c00, r0); r1 = wmma_bf16(a0, c10, r1);
  r2 = wmma_bf16(a0, c20, r2); r3 = wmma_bf16(a0, c30, r3);
  r0 = wmma_bf16(a1, c01, r0); r1 = wmma_bf16(a1, c11, r1);
  r2 = wmma_bf16(a1, c21, r2); r3 = wmma_bf16(a1, c31, r3);
  __bf16* D = out2 + (size_t)b * NN * CC + h * CHD;
  storeD_bf16(D, CC, n_tile,  0, r0, lane);
  storeD_bf16(D, CC, n_tile, 16, r1, lane);
  storeD_bf16(D, CC, n_tile, 32, r2, lane);
  storeD_bf16(D, CC, n_tile, 48, r3, lane);
}

// ---------------- Kernel 7: ffn GEMM  fout[n][o] = out2[n][c]*Wf[o][c] + bias[o] (fp32) ----------------
__global__ void k_ffn(const __bf16* __restrict__ out2, const __bf16* __restrict__ wf,
                      const float* __restrict__ bias, float* __restrict__ fout) {
  int wave = (blockIdx.x * blockDim.x + threadIdx.x) >> 5;
  int lane = threadIdx.x & 31;
  int b = wave >> 11;
  int rest = wave & 2047;
  int n_tile = (rest >> 3) << 4;
  int o_base = (rest & 7) << 6;
  const __bf16* A = out2 + (size_t)b * NN * CC;
  v8f acc[4] = {{}, {}, {}, {}};
  gemm16x64<CC>(A, CC, n_tile, wf, CC, o_base, lane, acc);
  int r0 = n_tile + ((lane >> 4) << 3);
  float* D = fout + (size_t)b * NN * CC;
  #pragma unroll
  for (int t = 0; t < 4; ++t) {
    int col = o_base + t * 16 + (lane & 15);
    float bv = bias[col];
    #pragma unroll
    for (int r = 0; r < 8; ++r)
      D[(size_t)(r0 + r) * CC + col] = acc[t][r] + bv;
  }
}

// ---------------- Kernel 8: RMSNorm over channel + residual -> d_out[b][c][n] ----------------
__global__ void k_norm_res(const float* __restrict__ fout, const float* __restrict__ g,
                           const float* __restrict__ x, float* __restrict__ out) {
  int wave = (blockIdx.x * blockDim.x + threadIdx.x) >> 5;  // one wave per pixel
  int lane = threadIdx.x & 31;
  int b = wave >> 12, n = wave & (NN - 1);
  const float* fp = fout + ((size_t)b * NN + n) * CC + lane * 16;
  float v[16]; float ss = 0.f;
  #pragma unroll
  for (int j = 0; j < 16; ++j) { float t = fp[j]; v[j] = t; ss += t * t; }
  ss = wave_sum(ss);
  float sc = SQRT_C / fmaxf(sqrtf(ss), 1e-12f);
  int c0 = lane * 16;
  #pragma unroll
  for (int j = 0; j < 16; ++j) {
    size_t idx = ((size_t)b * CC + c0 + j) * NN + n;
    out[idx] = v[j] * sc * g[c0 + j] + x[idx];
  }
}

// ---------------- Host launcher ----------------
extern "C" void kernel_launch(void* const* d_in, const int* in_sizes, int n_in,
                              void* d_out, int out_size, void* d_ws, size_t ws_size,
                              hipStream_t stream) {
  (void)in_sizes; (void)n_in; (void)out_size; (void)ws_size;
  const float* x      = (const float*)d_in[0];
  const float* norm_g = (const float*)d_in[1];
  const float* qkv_w  = (const float*)d_in[2];
  const float* ffn_w  = (const float*)d_in[3];
  const float* ffn_b  = (const float*)d_in[4];
  const float* ffn_ng = (const float*)d_in[5];
  float* outp = (float*)d_out;

  // workspace carve-up (bytes), 256-aligned
  char* w = (char*)d_ws;
  size_t off = 0;
  auto take = [&](size_t bytes) { char* p = w + off; off = (off + bytes + 255) & ~(size_t)255; return p; };
  __bf16* wq_bf = (__bf16*)take((size_t)3 * CC * CC * sizeof(__bf16));   // 1.5 MiB
  __bf16* wf_bf = (__bf16*)take((size_t)CC * CC * sizeof(__bf16));       // 0.5 MiB
  __bf16* xn    = (__bf16*)take((size_t)NB * NN * CC * sizeof(__bf16));  // 32 MiB
  __bf16* qraw  = (__bf16*)take((size_t)NB * NN * CC * sizeof(__bf16));  // 32 MiB
  __bf16* kv    = (__bf16*)take((size_t)NB * 1024 * NN * sizeof(__bf16)); // 64 MiB
  __bf16* attnt = (__bf16*)take((size_t)NB * NH * CHD * CHD * sizeof(__bf16)); // 0.5 MiB
  __bf16* out2  = xn;          // xn dead after k_gemm_kv
  float*  fout  = (float*)kv;  // kv dead after k_attn (same 64 MiB footprint)

  k_cvt_w     <<<3072, 256, 0, stream>>>(qkv_w, ffn_w, wq_bf, wf_bf);
  k_rmsnorm_t <<<4096, 256, 0, stream>>>(x, norm_g, xn);            // 32768 waves
  k_gemm_q    <<<2048, 256, 0, stream>>>(xn, wq_bf, qraw);          // 16384 waves
  k_gemm_kv   <<<4096, 256, 0, stream>>>(xn, wq_bf, kv);            // 32768 waves
  k_softmax_q <<<32768, 256, 0, stream>>>(qraw);                    // 262144 waves
  k_softmax_k <<<4096, 256, 0, stream>>>(kv);                       // block per (b,o)
  k_attn      <<<64, 512, 0, stream>>>(kv, attnt);                  // 16 waves per (b,h)
  k_out       <<<2048, 256, 0, stream>>>(qraw, attnt, out2);        // 16384 waves
  k_ffn       <<<2048, 256, 0, stream>>>(out2, wf_bf, ffn_b, fout); // 16384 waves
  k_norm_res  <<<4096, 256, 0, stream>>>(fout, ffn_ng, x, outp);    // 32768 waves
}
